// SparseFineGrainedContrastiveAlignment_13211319402735
// MI455X (gfx1250) — matrix operand order repaired and checked by reading, not compiled, not run on hardware
//
#include <hip/hip_runtime.h>
#include <hip/hip_bf16.h>
#include <math.h>

// ---------------- problem dims ----------------
#define BATCH 64
#define CDIM  256
#define HW    1024
#define SDIM  512
#define DDIM  256
#define TAUF  0.07f
#define COS_EPS 1e-8f
#define L1_EPS  1e-12f
#define MM_EPS  1e-8f

// ---------------- CDNA5 WMMA types ----------------
typedef __bf16 bf16_t;
typedef __attribute__((ext_vector_type(16))) __bf16 v16bf;
typedef __attribute__((ext_vector_type(8)))  float  v8f;
typedef __attribute__((ext_vector_type(4)))  int    v4i;

union BFrag { v16bf v; unsigned short u[16]; };

// ---------------- gfx1250 async global->LDS copy (ASYNCcnt path) ----------------
#if defined(__AMDGCN__) && __has_builtin(__builtin_amdgcn_global_load_async_to_lds_b128)
#define HAS_ASYNC_LDS 1
#else
#define HAS_ASYNC_LDS 0
#endif

typedef v4i __attribute__((address_space(1)))* gv4ip;  // global int4*
typedef v4i __attribute__((address_space(3)))* lv4ip;  // LDS int4*

__device__ __forceinline__ void copy16_g2l(const unsigned short* __restrict__ src,
                                           unsigned short* __restrict__ dst) {
#if HAS_ASYNC_LDS
  __builtin_amdgcn_global_load_async_to_lds_b128((gv4ip)src, (lv4ip)dst, 0, 0);
#else
  *reinterpret_cast<uint4*>(dst) = *reinterpret_cast<const uint4*>(src);
#endif
}

template <int N>
__device__ __forceinline__ void wait_async() {
#if HAS_ASYNC_LDS
#if __has_builtin(__builtin_amdgcn_s_wait_asynccnt)
  __builtin_amdgcn_s_wait_asynccnt(N);
#else
  asm volatile("s_wait_asynccnt %0" :: "i"(N) : "memory");
#endif
#endif
}

__device__ __forceinline__ unsigned short f2bfu(float f) {
  union { float f; unsigned u; } c; c.f = f;
  unsigned r = c.u + 0x7FFFu + ((c.u >> 16) & 1u);   // round-to-nearest-even
  return (unsigned short)(r >> 16);
}
__device__ __forceinline__ float bfu2f(unsigned short u) {
  union { unsigned u; float f; } c; c.u = ((unsigned)u) << 16;
  return c.f;
}
__device__ __forceinline__ void pack4_store(unsigned short* dst, float4 v) {
  uint2 u;
  u.x = (unsigned)f2bfu(v.x) | ((unsigned)f2bfu(v.y) << 16);
  u.y = (unsigned)f2bfu(v.z) | ((unsigned)f2bfu(v.w) << 16);
  *reinterpret_cast<uint2*>(dst) = u;
}
__device__ __forceinline__ v8f zero8() {
  v8f v;
#pragma unroll
  for (int i = 0; i < 8; ++i) v[i] = 0.0f;
  return v;
}

// A-fragment (16x32 bf16, MxK): lanes 0-15 -> M=lane, K-group 0; lanes 16-31 -> M=lane-16, K-group 8.
// VGPR 0..3: K = kg+2i, kg+2i+1 ; VGPR 4..7: K = 16+kg+2i, +1.  LDS tile is [rows][32] (k contiguous).
__device__ __forceinline__ v16bf load_a_frag(const unsigned short* As, int mBase, int lane) {
  BFrag f;
  const unsigned short* row = As + (mBase + (lane & 15)) * 32;
  int kg = (lane >> 4) * 8;
#pragma unroll
  for (int i = 0; i < 4; ++i) {
    int k = kg + 2 * i;
    f.u[2 * i]     = row[k];
    f.u[2 * i + 1] = row[k + 1];
  }
#pragma unroll
  for (int i = 0; i < 4; ++i) {
    int k = 16 + kg + 2 * i;
    f.u[8 + 2 * i] = row[k];
    f.u[9 + 2 * i] = row[k + 1];
  }
  return f.v;
}

// B-fragment (32x16 bf16, KxN): lanes 0-15 hold K=0..7,16..23 of column N=lane; lanes 16-31 hold K=8..15,24..31.
// LDS tile Bs is stored [n][32] (k contiguous per column).
__device__ __forceinline__ v16bf load_b_frag(const unsigned short* Bs, int nBase, int lane) {
  BFrag f;
  const unsigned short* col = Bs + (nBase + (lane & 15)) * 32;
  int kg = (lane >> 4) * 8;
#pragma unroll
  for (int j = 0; j < 16; ++j) f.u[j] = col[(j < 8) ? (kg + j) : (16 + kg + (j - 8))];
  return f.v;
}

// C/D 16x16 f32 layout: VGPR r: lanes 0-15 -> (M=r, N=lane); lanes 16-31 -> (M=8+r, N=lane-16)
__device__ __forceinline__ void store_acc_f32(float* C, size_t ldc, int mBase, int nBase,
                                              v8f acc, int lane) {
  int n  = lane & 15;
  int mo = (lane >> 4) * 8;
#pragma unroll
  for (int r = 0; r < 8; ++r)
    C[(size_t)(mBase + mo + r) * ldc + (nBase + n)] = acc[r];
}

// one wave: 32x64 micro-tile MAC over a K=32 chunk staged in LDS (2 A-frags x 4 B-frags -> 8 WMMA)
__device__ __forceinline__ void wmma_mac_32x64(const unsigned short* As, const unsigned short* Bs,
                                               int waveM, int waveN, int lane, v8f acc[2][4]) {
  v16bf a[2], b[4];
  a[0] = load_a_frag(As, waveM,      lane);
  a[1] = load_a_frag(As, waveM + 16, lane);
#pragma unroll
  for (int j = 0; j < 4; ++j) b[j] = load_b_frag(Bs, waveN + 16 * j, lane);
#pragma unroll
  for (int i = 0; i < 2; ++i)
#pragma unroll
    for (int j = 0; j < 4; ++j)
      acc[i][j] = __builtin_amdgcn_wmma_f32_16x16x32_bf16(false, a[i], false, b[j],
                                                          (short)0, acc[i][j], false, false);
}

// ================= adapter GEMM: out_bf16[b][m][d] = A[b] (MxK) @ W^T (KxD) + bias =================
// transA==1: A element (m,k) = Ain[b*K*M + k*M + m]  (image path: [b][c][hw])
// transA==0: A element (m,k) = Ain[b*M*K + m*K + k]  (token path: [b][s][d])
// K = CDIM, N-tile 128, M-tile 128, 256 threads. f32 input -> convert to bf16 while staging.
__global__ __launch_bounds__(256) void k_adapter(const float* __restrict__ Ain,
                                                 const float* __restrict__ W,
                                                 const float* __restrict__ bias,
                                                 unsigned short* __restrict__ outp,
                                                 int M, int transA) {
  __shared__ unsigned short As[128 * 32];
  __shared__ unsigned short Bs[128 * 32];
  const int tid = threadIdx.x, lane = tid & 31, wave = tid >> 5;
  const int waveM = (wave & 3) * 32, waveN = (wave >> 2) * 64;
  const int b = blockIdx.z;
  const int mBase = blockIdx.x * 128;
  const int nBase = blockIdx.y * 128;
  const size_t batA = (size_t)M * CDIM;

  v8f acc[2][4];
#pragma unroll
  for (int i = 0; i < 2; ++i)
#pragma unroll
    for (int j = 0; j < 4; ++j) acc[i][j] = zero8();

  for (int k0 = 0; k0 < CDIM; k0 += 32) {
    if (transA) {
      // contiguous in m: float4 along m, 4 scalar LDS stores
#pragma unroll
      for (int j = 0; j < 4; ++j) {
        int i = j * 256 + tid;
        int k = i >> 5, mc = (i & 31) * 4;
        float4 v = *reinterpret_cast<const float4*>(
            &Ain[(size_t)b * batA + (size_t)(k0 + k) * M + (mBase + mc)]);
        As[(mc + 0) * 32 + k] = f2bfu(v.x);
        As[(mc + 1) * 32 + k] = f2bfu(v.y);
        As[(mc + 2) * 32 + k] = f2bfu(v.z);
        As[(mc + 3) * 32 + k] = f2bfu(v.w);
      }
    } else {
      // contiguous in k: float4 along k, packed uint2 LDS store
#pragma unroll
      for (int j = 0; j < 4; ++j) {
        int i = j * 256 + tid;
        int m = i >> 3, kc = (i & 7) * 4;
        float4 v = *reinterpret_cast<const float4*>(
            &Ain[(size_t)b * batA + (size_t)(mBase + m) * CDIM + (k0 + kc)]);
        pack4_store(&As[m * 32 + kc], v);
      }
    }
    // B(k,n) = W[n][k] -> Bs[n][k], contiguous in k
#pragma unroll
    for (int j = 0; j < 4; ++j) {
      int i = j * 256 + tid;
      int n = i >> 3, kc = (i & 7) * 4;
      float4 v = *reinterpret_cast<const float4*>(
          &W[(size_t)(nBase + n) * CDIM + (k0 + kc)]);
      pack4_store(&Bs[n * 32 + kc], v);
    }
    __syncthreads();
    wmma_mac_32x64(As, Bs, waveM, waveN, lane, acc);
    __syncthreads();
  }

  int n = lane & 15, mo = (lane >> 4) * 8;
#pragma unroll
  for (int sm = 0; sm < 2; ++sm)
#pragma unroll
    for (int sn = 0; sn < 4; ++sn) {
      int gn = nBase + waveN + sn * 16 + n;
      float bb = bias[gn];
#pragma unroll
      for (int r = 0; r < 8; ++r) {
        int gm = mBase + waveM + sm * 16 + mo + r;
        outp[((size_t)b * M + gm) * DDIM + gn] = f2bfu(acc[sm][sn][r] + bb);
      }
    }
}

// ================= generic batched bf16 GEMM: C[b] (MxN,f32) = A[b] @ Bt[b]^T ====================
// A: row-major [M][K] bf16.  Bt: row-major [N][K] bf16 (operand B transposed).
// 128x128 block tile, double-buffered async global->LDS staging (4 async instructions per chunk).
__device__ __forceinline__ void stage_chunk(const unsigned short* __restrict__ Ab,
                                            const unsigned short* __restrict__ Bb,
                                            int K, int mBase, int nBase, int k0, int tid,
                                            unsigned short* As, unsigned short* Bs) {
#pragma unroll
  for (int j = 0; j < 2; ++j) {
    int i = j * 256 + tid;
    int m = i >> 2, kk = (i & 3) * 8;
    copy16_g2l(Ab + (size_t)(mBase + m) * K + (k0 + kk), &As[m * 32 + kk]);
  }
#pragma unroll
  for (int j = 0; j < 2; ++j) {
    int i = j * 256 + tid;
    int n = i >> 2, kk = (i & 3) * 8;
    copy16_g2l(Bb + (size_t)(nBase + n) * K + (k0 + kk), &Bs[n * 32 + kk]);
  }
}

__global__ __launch_bounds__(256) void k_bgemm(const unsigned short* __restrict__ Asrc,
                                               const unsigned short* __restrict__ Btsrc,
                                               float* __restrict__ C,
                                               int M, int N, int K) {
  __shared__ unsigned short As[2][128 * 32];
  __shared__ unsigned short Bs[2][128 * 32];
  const int tid = threadIdx.x, lane = tid & 31, wave = tid >> 5;
  const int waveM = (wave & 3) * 32, waveN = (wave >> 2) * 64;
  const int b = blockIdx.z;
  const int mBase = blockIdx.x * 128;
  const int nBase = blockIdx.y * 128;
  const unsigned short* Ab = Asrc + (size_t)b * (size_t)M * K;
  const unsigned short* Bb = Btsrc + (size_t)b * (size_t)N * K;

  v8f acc[2][4];
#pragma unroll
  for (int i = 0; i < 2; ++i)
#pragma unroll
    for (int j = 0; j < 4; ++j) acc[i][j] = zero8();

  stage_chunk(Ab, Bb, K, mBase, nBase, 0, tid, As[0], Bs[0]);
  int cur = 0;
  for (int k0 = 0; k0 < K; k0 += 32) {
    int nxt = cur ^ 1;
    if (k0 + 32 < K) {
      stage_chunk(Ab, Bb, K, mBase, nBase, k0 + 32, tid, As[nxt], Bs[nxt]);
      wait_async<4>();   // only the 4 just-issued may remain outstanding -> chunk k0 is in LDS
    } else {
      wait_async<0>();
    }
    __syncthreads();
    wmma_mac_32x64(As[cur], Bs[cur], waveM, waveN, lane, acc);
    __syncthreads();
    cur = nxt;
  }

  float* Cb = C + (size_t)b * M * N;
#pragma unroll
  for (int sm = 0; sm < 2; ++sm)
#pragma unroll
    for (int sn = 0; sn < 4; ++sn)
      store_acc_f32(Cb, (size_t)N, mBase + waveM + sm * 16, nBase + waveN + sn * 16,
                    acc[sm][sn], lane);
}

// ================= bf16 tile transpose: X[b][HW][DDIM] -> Y[b][DDIM][HW] =========================
__global__ __launch_bounds__(256) void k_transpose(const unsigned short* __restrict__ X,
                                                   unsigned short* __restrict__ Y) {
  __shared__ unsigned short tile[32][33];
  int b = blockIdx.z;
  int p0 = blockIdx.x * 32, d0 = blockIdx.y * 32;
  int t = threadIdx.x;
#pragma unroll
  for (int j = 0; j < 4; ++j) {
    int i = j * 256 + t;
    int p = i >> 5, d = i & 31;
    tile[p][d] = X[((size_t)b * HW + p0 + p) * DDIM + d0 + d];
  }
  __syncthreads();
#pragma unroll
  for (int j = 0; j < 4; ++j) {
    int i = j * 256 + t;
    int d = i >> 5, p = i & 31;
    Y[((size_t)b * DDIM + d0 + d) * HW + p0 + p] = tile[p][d];
  }
}

// ================= mean pool over rows: mean[b][d] = sum_r X[b][r][d] / rows =====================
__global__ __launch_bounds__(256) void k_pool(const unsigned short* __restrict__ X,
                                              float* __restrict__ meanOut, int rows) {
  int b = blockIdx.x, d = threadIdx.x;
  const unsigned short* Xb = X + (size_t)b * rows * DDIM;
  float s = 0.0f;
  for (int r = 0; r < rows; ++r) s += bfu2f(Xb[(size_t)r * DDIM + d]);
  meanOut[(size_t)b * DDIM + d] = s / (float)rows;
}

// ================= g = l2n(mean @ W^T + bias) (second adapter application) =======================
__global__ __launch_bounds__(256) void k_global_embed(const float* __restrict__ meanIn,
                                                      const float* __restrict__ W,
                                                      const float* __restrict__ bias,
                                                      float* __restrict__ gn) {
  __shared__ float mrow[DDIM];
  __shared__ float red[256];
  int b = blockIdx.x, d = threadIdx.x;
  mrow[d] = meanIn[(size_t)b * DDIM + d];
  __syncthreads();
  float acc = bias[d];
  for (int c = 0; c < DDIM; ++c) acc += mrow[c] * W[(size_t)d * DDIM + c];
  red[d] = acc * acc;
  __syncthreads();
  for (int s = 128; s > 0; s >>= 1) { if (d < s) red[d] += red[d + s]; __syncthreads(); }
  float nrm = fmaxf(sqrtf(red[0]), COS_EPS);
  gn[(size_t)b * DDIM + d] = acc / nrm;
}

// ================= global symmetric CE over 64x64 sim ============================================
__global__ __launch_bounds__(256) void k_gloss(const float* __restrict__ gi,
                                               const float* __restrict__ gt,
                                               float* __restrict__ out) {
  __shared__ float sim[BATCH * BATCH];
  __shared__ float red[256];
  int t = threadIdx.x;
  for (int i = t; i < BATCH * BATCH; i += 256) {
    int r = i >> 6, c = i & 63;
    float a = 0.0f;
    for (int k = 0; k < DDIM; ++k) a += gi[(size_t)r * DDIM + k] * gt[(size_t)c * DDIM + k];
    sim[i] = a / TAUF;
  }
  __syncthreads();
  float l = 0.0f;
  if (t < BATCH) {
    float m = -1e30f;
    for (int j = 0; j < BATCH; ++j) m = fmaxf(m, sim[t * BATCH + j]);
    float se = 0.0f;
    for (int j = 0; j < BATCH; ++j) se += expf(sim[t * BATCH + j] - m);
    float lr = m + logf(se);
    float m2 = -1e30f;
    for (int j = 0; j < BATCH; ++j) m2 = fmaxf(m2, sim[j * BATCH + t]);
    float se2 = 0.0f;
    for (int j = 0; j < BATCH; ++j) se2 += expf(sim[j * BATCH + t] - m2);
    float lc = m2 + logf(se2);
    float dg = sim[t * BATCH + t];
    l = 0.5f * ((lr - dg) + (lc - dg)) / (float)BATCH;
  }
  red[t] = l;
  __syncthreads();
  for (int s = 128; s > 0; s >>= 1) { if (t < s) red[t] += red[t + s]; __syncthreads(); }
  if (t == 0) atomicAdd(out, red[0]);
}

// ================= column stats + sparse L1 weights, TRANSPOSED output ===========================
__global__ __launch_bounds__(256) void k_weights(const float* __restrict__ sim,
                                                 unsigned short* __restrict__ wT) {
  __shared__ unsigned short tile[256][33];
  int b = blockIdx.y;
  int sBase = blockIdx.x * 256;
  int t = threadIdx.x;
  int s = sBase + t;
  const float* simb = sim + (size_t)b * HW * SDIM;

  float mn = 1e30f, mx = -1e30f;
  for (int p = 0; p < HW; ++p) {
    float v = simb[(size_t)p * SDIM + s];
    mn = fminf(mn, v); mx = fmaxf(mx, v);
  }
  float inv = 1.0f / (mx - mn + MM_EPS);
  const float sigma = 1.0f / (float)HW;
  float sum = 0.0f;
  for (int p = 0; p < HW; ++p) {
    float w = (simb[(size_t)p * SDIM + s] - mn) * inv;
    w = (w < sigma) ? 0.0f : w;
    sum += w;
  }
  float dinv = 1.0f / fmaxf(sum, L1_EPS);

  for (int p0 = 0; p0 < HW; p0 += 32) {
#pragma unroll
    for (int pp = 0; pp < 32; ++pp) {
      float w = (simb[(size_t)(p0 + pp) * SDIM + s] - mn) * inv;
      w = (w < sigma) ? 0.0f : w;
      tile[t][pp] = f2bfu(w * dinv);
    }
    __syncthreads();
#pragma unroll
    for (int j = 0; j < 32; ++j) {
      int i = j * 256 + t;
      int p = i & 31, srow = i >> 5;
      wT[((size_t)b * SDIM + sBase + srow) * HW + p0 + p] = tile[srow][p];
    }
    __syncthreads();
  }
}

// ================= row l2-normalize (f32 in -> bf16 out) =========================================
__global__ __launch_bounds__(256) void k_norm_f32(const float* __restrict__ X,
                                                  unsigned short* __restrict__ Y) {
  __shared__ float red[256];
  size_t row = blockIdx.x;
  int d = threadIdx.x;
  float v = X[row * DDIM + d];
  red[d] = v * v;
  __syncthreads();
  for (int s = 128; s > 0; s >>= 1) { if (d < s) red[d] += red[d + s]; __syncthreads(); }
  float nrm = fmaxf(sqrtf(red[0]), COS_EPS);
  Y[row * DDIM + d] = f2bfu(v / nrm);
}

// ================= row l2-normalize (bf16 in -> bf16 out) ========================================
__global__ __launch_bounds__(256) void k_norm_b16(const unsigned short* __restrict__ X,
                                                  unsigned short* __restrict__ Y) {
  __shared__ float red[256];
  size_t row = blockIdx.x;
  int d = threadIdx.x;
  float v = bfu2f(X[row * DDIM + d]);
  red[d] = v * v;
  __syncthreads();
  for (int s = 128; s > 0; s >>= 1) { if (d < s) red[d] += red[d + s]; __syncthreads(); }
  float nrm = fmaxf(sqrtf(red[0]), COS_EPS);
  Y[row * DDIM + d] = f2bfu(v / nrm);
}

// ================= fine-grained symmetric CE: block per (i, b) ===================================
__global__ __launch_bounds__(256) void k_floss(const float* __restrict__ fsim,
                                               float* __restrict__ out) {
  __shared__ float red[256];
  int b = blockIdx.y, i = blockIdx.x, t = threadIdx.x;
  const float* F = fsim + (size_t)b * SDIM * SDIM;
  const float itau = 1.0f / TAUF;

  float v0 = F[(size_t)i * SDIM + t] * itau;
  float v1 = F[(size_t)i * SDIM + t + 256] * itau;
  red[t] = fmaxf(v0, v1);
  __syncthreads();
  for (int s = 128; s > 0; s >>= 1) { if (t < s) red[t] = fmaxf(red[t], red[t + s]); __syncthreads(); }
  float rm = red[0];
  __syncthreads();
  red[t] = expf(v0 - rm) + expf(v1 - rm);
  __syncthreads();
  for (int s = 128; s > 0; s >>= 1) { if (t < s) red[t] += red[t + s]; __syncthreads(); }
  float lr = rm + logf(red[0]);
  __syncthreads();

  float c0 = F[(size_t)t * SDIM + i] * itau;
  float c1 = F[(size_t)(t + 256) * SDIM + i] * itau;
  red[t] = fmaxf(c0, c1);
  __syncthreads();
  for (int s = 128; s > 0; s >>= 1) { if (t < s) red[t] = fmaxf(red[t], red[t + s]); __syncthreads(); }
  float cm = red[0];
  __syncthreads();
  red[t] = expf(c0 - cm) + expf(c1 - cm);
  __syncthreads();
  for (int s = 128; s > 0; s >>= 1) { if (t < s) red[t] += red[t + s]; __syncthreads(); }
  float lc = cm + logf(red[0]);

  if (t == 0) {
    float dg = F[(size_t)i * SDIM + i] * itau;
    atomicAdd(out, 0.5f * ((lr - dg) + (lc - dg)) / ((float)BATCH * (float)SDIM));
  }
}

// ================================== host orchestration ===========================================
extern "C" void kernel_launch(void* const* d_in, const int* in_sizes, int n_in,
                              void* d_out, int out_size, void* d_ws, size_t ws_size,
                              hipStream_t stream) {
  (void)in_sizes; (void)n_in; (void)ws_size;
  const float* img = (const float*)d_in[0];  // [B, C, H, W] = [64, 256, 1024]
  const float* tok = (const float*)d_in[1];  // [B, S, D]    = [64, 512, 256]
  const float* Wv  = (const float*)d_in[2];  // [D, D]
  const float* bv  = (const float*)d_in[3];  // [D]
  const float* Wt  = (const float*)d_in[4];  // [D, D]
  const float* bt  = (const float*)d_in[5];  // [D]
  float* out = (float*)d_out;

  size_t off = 0;
  char* base = (char*)d_ws;
  auto alloc = [&](size_t bytes) -> void* {
    void* r = base + off;
    off = (off + bytes + 255) & ~(size_t)255;
    return r;
  };
  unsigned short* patchesB = (unsigned short*)alloc((size_t)BATCH * HW * DDIM * 2);     // 32 MB
  unsigned short* patchesT = (unsigned short*)alloc((size_t)BATCH * DDIM * HW * 2);     // 32 MB
  unsigned short* tokensB  = (unsigned short*)alloc((size_t)BATCH * SDIM * DDIM * 2);   // 16 MB
  float*          simbuf   = (float*)alloc((size_t)BATCH * HW * SDIM * 4);              // 128 MB (reused for f_sim)
  unsigned short* weightsT = (unsigned short*)alloc((size_t)BATCH * SDIM * HW * 2);     // 64 MB
  float*          lgve     = (float*)alloc((size_t)BATCH * SDIM * DDIM * 4);            // 32 MB
  unsigned short* lgveN    = (unsigned short*)alloc((size_t)BATCH * SDIM * DDIM * 2);   // 16 MB
  unsigned short* tokensN  = (unsigned short*)alloc((size_t)BATCH * SDIM * DDIM * 2);   // 16 MB
  float* meanp = (float*)alloc((size_t)BATCH * DDIM * 4);
  float* meant = (float*)alloc((size_t)BATCH * DDIM * 4);
  float* giN   = (float*)alloc((size_t)BATCH * DDIM * 4);
  float* gtN   = (float*)alloc((size_t)BATCH * DDIM * 4);
  float* fsim  = simbuf;  // alias: sim dead after k_weights

  (void)hipMemsetAsync(d_out, 0, (size_t)out_size * sizeof(float), stream);

  // adapters (WMMA bf16): patches[b,p,d], tokens[b,s,d]
  k_adapter<<<dim3(HW / 128, DDIM / 128, BATCH), 256, 0, stream>>>(img, Wv, bv, patchesB, HW, 1);
  k_adapter<<<dim3(SDIM / 128, DDIM / 128, BATCH), 256, 0, stream>>>(tok, Wt, bt, tokensB, SDIM, 0);

  // patchesT[b,d,p] for the lgve GEMM's B operand
  k_transpose<<<dim3(HW / 32, DDIM / 32, BATCH), 256, 0, stream>>>(patchesB, patchesT);

  // global path
  k_pool<<<BATCH, 256, 0, stream>>>(patchesB, meanp, HW);
  k_pool<<<BATCH, 256, 0, stream>>>(tokensB, meant, SDIM);
  k_global_embed<<<BATCH, 256, 0, stream>>>(meanp, Wv, bv, giN);
  k_global_embed<<<BATCH, 256, 0, stream>>>(meant, Wt, bt, gtN);
  k_gloss<<<1, 256, 0, stream>>>(giN, gtN, out);

  // sim[b,p,s] = patches @ tokens^T  (WMMA bf16; Bt = tokens [s][d])
  k_bgemm<<<dim3(HW / 128, SDIM / 128, BATCH), 256, 0, stream>>>(
      patchesB, tokensB, simbuf, HW, SDIM, DDIM);

  // min-max renorm + threshold + L1 normalize over p -> bf16 weightsT[b,s,p]
  k_weights<<<dim3(SDIM / 256, BATCH), 256, 0, stream>>>(simbuf, weightsT);

  // lgve[b,s,d] = weights^T @ patches  (A = weightsT [s][p], Bt = patchesT [d][p], K = HW)
  k_bgemm<<<dim3(SDIM / 128, DDIM / 128, BATCH), 256, 0, stream>>>(
      weightsT, patchesT, lgve, SDIM, DDIM, HW);

  // row l2 norms -> bf16
  k_norm_f32<<<BATCH * SDIM, 256, 0, stream>>>(lgve, lgveN);
  k_norm_b16<<<BATCH * SDIM, 256, 0, stream>>>(tokensB, tokensN);

  // f_sim[b,i,j] = l2n(lgve) @ l2n(tokens)^T  (A = lgveN [i][d], Bt = tokensN [j][d])
  k_bgemm<<<dim3(SDIM / 128, SDIM / 128, BATCH), 256, 0, stream>>>(
      lgveN, tokensN, fsim, SDIM, SDIM, DDIM);

  // fine-grained symmetric CE
  k_floss<<<dim3(SDIM, BATCH), 256, 0, stream>>>(fsim, out);
}
